// So3kratesBlock_74328704024953
// MI455X (gfx1250) — matrix operand order repaired and checked by reading, not compiled, not run on hardware
//
#include <hip/hip_runtime.h>
#include <hip/hip_bf16.h>

typedef __attribute__((ext_vector_type(16))) _Float16 v16h;
typedef __attribute__((ext_vector_type(8)))  _Float16 v8h;
typedef __attribute__((ext_vector_type(4)))  _Float16 v4h;
typedef __attribute__((ext_vector_type(8)))  float    v8f;

#define NNODES 20000
#define NEDGES 640000
#define CDIM   128
#define HIDD   64
#define RBFN   32
#define SHD    15
#define CATS   132      // f32 h-buffer row stride
#define CATP   160      // f16 zero-padded concat stride (K and N pad for ib block)
#define CUTR   5.0f
#define INV_AVG (1.0f/32.0f)

__device__ __forceinline__ float silu_f(float x) { return x / (1.0f + __expf(-x)); }

// ---------------- small helpers ----------------
__global__ void k_zero(float* __restrict__ p, int n) {
  int i = blockIdx.x * blockDim.x + threadIdx.x;
  if (i < n) p[i] = 0.0f;
}

__global__ void k_cvt(const float* __restrict__ a, _Float16* __restrict__ b, int n) {
  int i = blockIdx.x * blockDim.x + threadIdx.x;
  if (i < n) b[i] = (_Float16)a[i];
}

// W (K x N, f32) -> Wt (Np x Kp, f16) transposed + zero padded
__global__ void k_wt(const float* __restrict__ W, _Float16* __restrict__ Wt,
                     int K, int N, int Kp, int Np) {
  int i = blockIdx.x * blockDim.x + threadIdx.x;
  if (i >= Np * Kp) return;
  int n = i / Kp, k = i % Kp;
  Wt[i] = (n < N && k < K) ? (_Float16)W[(size_t)k * N + n] : (_Float16)0.f;
}

__global__ void k_embed(const int* __restrict__ spec, const float* __restrict__ emb,
                        float* __restrict__ nf) {
  int i = blockIdx.x * blockDim.x + threadIdx.x;
  if (i >= NNODES * CDIM) return;
  int n = i >> 7, c = i & 127;
  nf[i] = emb[spec[n] * CDIM + c];
}

// Bessel RBF (f16, E x 32) + real spherical harmonics l=1..3 (f32, E x 15)
__global__ void k_edge_geom(const float* __restrict__ vec, const float* __restrict__ dist,
                            _Float16* __restrict__ feats, float* __restrict__ sh) {
  int e = blockIdx.x * blockDim.x + threadIdx.x;
  if (e >= NEDGES) return;
  float d  = dist[e];
  float s0 = sqrtf(2.0f / CUTR) / d;
  float arg = 3.14159265358979f * d / CUTR;
  _Float16* fo = feats + (size_t)e * RBFN;
  #pragma unroll
  for (int n = 1; n <= RBFN; ++n) fo[n - 1] = (_Float16)(s0 * __sinf(arg * (float)n));
  float x = vec[e * 3 + 0], y = vec[e * 3 + 1], z = vec[e * 3 + 2];
  float inv = 1.0f / (sqrtf(x * x + y * y + z * z) + 1e-12f);
  x *= inv; y *= inv; z *= inv;
  float x2 = x * x, y2 = y * y, z2 = z * z;
  float* o = sh + (size_t)e * SHD;
  const float s3 = 1.7320508f, s15 = 3.8729833f, s5h = 1.1180340f;
  o[0] = s3 * x; o[1] = s3 * y; o[2] = s3 * z;
  o[3] = s15 * x * y; o[4] = s15 * y * z; o[5] = s5h * (3.f * z2 - 1.f);
  o[6] = s15 * x * z; o[7] = 0.5f * s15 * (x2 - y2);
  const float a3 = 2.0916500f, b3 = 10.246951f, c3 = 1.6201851f, d3 = 1.3228757f;
  o[8]  = a3 * y * (3.f * x2 - y2);
  o[9]  = b3 * x * y * z;
  o[10] = c3 * y * (5.f * z2 - 1.f);
  o[11] = d3 * z * (5.f * z2 - 3.f);
  o[12] = c3 * x * (5.f * z2 - 1.f);
  o[13] = 0.5f * b3 * z * (x2 - y2);
  o[14] = a3 * x * (x2 - 3.f * y2);
}

// chi_scalar = deg_norms(chi[snd]-chi[rcv]) -> f16 E x 32, zero padded (WMMA A input)
__global__ void k_chi_scalar(const float* __restrict__ chi, const int* __restrict__ snd,
                             const int* __restrict__ rcv, _Float16* __restrict__ out) {
  int e = blockIdx.x * blockDim.x + threadIdx.x;
  if (e >= NEDGES) return;
  const float* a = chi + (size_t)snd[e] * SHD;
  const float* b = chi + (size_t)rcv[e] * SHD;
  float acc0 = 1e-12f, acc1 = 1e-12f, acc2 = 1e-12f;
  #pragma unroll
  for (int s = 0; s < 3;  ++s) { float d = a[s] - b[s]; acc0 += d * d; }
  #pragma unroll
  for (int s = 3; s < 8;  ++s) { float d = a[s] - b[s]; acc1 += d * d; }
  #pragma unroll
  for (int s = 8; s < 15; ++s) { float d = a[s] - b[s]; acc2 += d * d; }
  _Float16* o = out + (size_t)e * 32;
  o[0] = (_Float16)sqrtf(acc0);
  o[1] = (_Float16)sqrtf(acc1);
  o[2] = (_Float16)sqrtf(acc2);
  #pragma unroll
  for (int s = 3; s < 32; ++s) o[s] = (_Float16)0.f;
}

// ---------------- WMMA GEMM, branch-free inner loop ----------------
// A: f16 row-major (M x lda), M%16==0, K%32==0 (zero padded).
// Bt: f16 transposed (Npad x ldb), ldb=Kp, zero padded.
// Per-lane fragment loads are 16B vector loads (ISA 7.12.2 layouts):
//   A elem i -> k = 8*half + (i<8?i:i+8)  => two contiguous v8h runs
//   B elem i -> k = 8*half + (i<8?i:i+8)  => two contiguous v8h runs down Bt row
// flags: 1=silu  2=accumulate into C(f32)  4=bias  8=store f16
__global__ void k_gemm(const _Float16* __restrict__ A, int lda,
                       const _Float16* __restrict__ Bt, int ldb,
                       const float* __restrict__ bias,
                       float* __restrict__ Cm, int ldc,
                       int M, int Nreal, int K, int flags) {
  const int lane = threadIdx.x;
  const int half = lane >> 4, rm = lane & 15;
  const int m0 = (blockIdx.y * 4 + threadIdx.y) * 16;
  const int n0 = blockIdx.x * 32;          // two 16-wide N tiles per wave
  if (m0 >= M) return;
  const _Float16* Arow = A  + (size_t)(m0 + rm) * lda;
  const _Float16* B0   = Bt + (size_t)(n0 + rm) * ldb;
  const _Float16* B1   = Bt + (size_t)(n0 + 16 + rm) * ldb;
  v8f acc0 = {0.f,0.f,0.f,0.f,0.f,0.f,0.f,0.f};
  v8f acc1 = {0.f,0.f,0.f,0.f,0.f,0.f,0.f,0.f};
  for (int k0 = 0; k0 < K; k0 += 32) {
    v8h alo = *(const v8h*)(Arow + k0 + 8 * half);
    v8h ahi = *(const v8h*)(Arow + k0 + 16 + 8 * half);
    v16h a = __builtin_shufflevector(alo, ahi, 0,1,2,3,4,5,6,7,8,9,10,11,12,13,14,15);
    v8h b0l = *(const v8h*)(B0 + k0 + 8 * half);
    v8h b0h = *(const v8h*)(B0 + k0 + 16 + 8 * half);
    v16h b0 = __builtin_shufflevector(b0l, b0h, 0,1,2,3,4,5,6,7,8,9,10,11,12,13,14,15);
    acc0 = __builtin_amdgcn_wmma_f32_16x16x32_f16(false, a, false, b0, (short)0, acc0,
                                                  false, false);
    v8h b1l = *(const v8h*)(B1 + k0 + 8 * half);
    v8h b1h = *(const v8h*)(B1 + k0 + 16 + 8 * half);
    v16h b1 = __builtin_shufflevector(b1l, b1h, 0,1,2,3,4,5,6,7,8,9,10,11,12,13,14,15);
    acc1 = __builtin_amdgcn_wmma_f32_16x16x32_f16(false, a, false, b1, (short)0, acc1,
                                                  false, false);
  }
  #pragma unroll
  for (int t = 0; t < 2; ++t) {
    int col = n0 + t * 16 + rm;
    if (col >= Nreal) continue;
    v8f acc = t ? acc1 : acc0;
    float bi = (flags & 4) ? bias[col] : 0.f;
    #pragma unroll
    for (int r = 0; r < 8; ++r) {
      int m = m0 + r + 8 * half;
      float v = acc[r] + bi;
      if (flags & 1) v = silu_f(v);
      size_t idx = (size_t)m * ldc + col;
      if (flags & 8) {
        ((_Float16*)Cm)[idx] = (_Float16)v;
      } else {
        if (flags & 2) v += Cm[idx];
        Cm[idx] = v;
      }
    }
  }
}

// ---------------- fused edge MLP: out16 = (X@W1+b1 |> silu) @ W2 + b2 ----------------
// X: f16 E x 32 (zero padded); W1t: f16 64 x 32; W2t: f16 128 x 64. mul: out *= result.
__global__ void k_edge_mlp(const _Float16* __restrict__ X,
                           const _Float16* __restrict__ W1t, const float* __restrict__ b1,
                           const _Float16* __restrict__ W2t, const float* __restrict__ b2,
                           _Float16* __restrict__ out, int mul) {
  __shared__ _Float16 hid[4][16][72];   // 72*2B row stride = 144B (16B aligned)
  const int lane = threadIdx.x, w = threadIdx.y;
  const int half = lane >> 4, rm = lane & 15;
  const int e0 = (blockIdx.x * 4 + w) * 16;

  const _Float16* Xrow = X + (size_t)(e0 + rm) * 32;
  v8h alo = *(const v8h*)(Xrow + 8 * half);
  v8h ahi = *(const v8h*)(Xrow + 16 + 8 * half);
  v16h a = __builtin_shufflevector(alo, ahi, 0,1,2,3,4,5,6,7,8,9,10,11,12,13,14,15);
  #pragma unroll
  for (int j = 0; j < 4; ++j) {
    const _Float16* Brow = W1t + (size_t)(j * 16 + rm) * 32 + 8 * half;
    v8h bl = *(const v8h*)(Brow);
    v8h bh = *(const v8h*)(Brow + 16);
    v16h b = __builtin_shufflevector(bl, bh, 0,1,2,3,4,5,6,7,8,9,10,11,12,13,14,15);
    v8f h0 = {0.f,0.f,0.f,0.f,0.f,0.f,0.f,0.f};
    h0 = __builtin_amdgcn_wmma_f32_16x16x32_f16(false, a, false, b, (short)0, h0,
                                                false, false);
    #pragma unroll
    for (int r = 0; r < 8; ++r)
      hid[w][r + 8 * half][j * 16 + rm] = (_Float16)silu_f(h0[r] + b1[j * 16 + rm]);
  }
  __syncthreads();

  v8f acc[8];
  #pragma unroll
  for (int j = 0; j < 8; ++j) acc[j] = (v8f){0.f,0.f,0.f,0.f,0.f,0.f,0.f,0.f};
  #pragma unroll
  for (int s = 0; s < 2; ++s) {
    const _Float16* Hrow = &hid[w][rm][s * 32 + 8 * half];
    v8h hlo = *(const v8h*)(Hrow);
    v8h hhi = *(const v8h*)(Hrow + 16);
    v16h a2 = __builtin_shufflevector(hlo, hhi, 0,1,2,3,4,5,6,7,8,9,10,11,12,13,14,15);
    #pragma unroll
    for (int j = 0; j < 8; ++j) {
      const _Float16* Brow = W2t + (size_t)(j * 16 + rm) * 64 + s * 32 + 8 * half;
      v8h bl = *(const v8h*)(Brow);
      v8h bh = *(const v8h*)(Brow + 16);
      v16h b = __builtin_shufflevector(bl, bh, 0,1,2,3,4,5,6,7,8,9,10,11,12,13,14,15);
      acc[j] = __builtin_amdgcn_wmma_f32_16x16x32_f16(false, a2, false, b, (short)0,
                                                      acc[j], false, false);
    }
  }
  #pragma unroll
  for (int j = 0; j < 8; ++j) {
    int n = j * 16 + rm;
    float bi = b2[n];
    #pragma unroll
    for (int r = 0; r < 8; ++r) {
      int e = e0 + r + 8 * half;      // NEDGES % 16 == 0: no guard needed
      size_t idx = (size_t)e * CDIM + n;
      float v = acc[j][r] + bi;
      out[idx] = mul ? (_Float16)((float)out[idx] * v) : (_Float16)v;
    }
  }
}

// ---------------- attention edges (f16 operands, f32 atomics) ----------------
__global__ void k_attn(const _Float16* __restrict__ q, const _Float16* __restrict__ kk,
                       const _Float16* __restrict__ v, const _Float16* __restrict__ w,
                       const float* __restrict__ cut, const int* __restrict__ snd,
                       const int* __restrict__ rcv, float* __restrict__ agg) {
  int e = blockIdx.x * blockDim.y + threadIdx.y;
  if (e >= NEDGES) return;
  int lane = threadIdx.x;
  int s = snd[e], r = rcv[e];
  int c0 = lane * 4;
  const v4h qv = *(const v4h*)(q  + (size_t)r * CDIM + c0);
  const v4h kv = *(const v4h*)(kk + (size_t)s * CDIM + c0);
  const v4h wv = *(const v4h*)(w  + (size_t)e * CDIM + c0);
  float dot = 0.f;
  #pragma unroll
  for (int i = 0; i < 4; ++i) dot += (float)qv[i] * (float)wv[i] * (float)kv[i];
  dot += __shfl_xor(dot, 1, 32);   // head group = 8 lanes (DH=32)
  dot += __shfl_xor(dot, 2, 32);
  dot += __shfl_xor(dot, 4, 32);
  float alpha = dot * 0.17677670f * cut[e] * INV_AVG;  // 1/sqrt(DH), fold 1/AVG_NEIGH
  const v4h vv = *(const v4h*)(v + (size_t)s * CDIM + c0);
  #pragma unroll
  for (int i = 0; i < 4; ++i)
    atomicAdd(&agg[(size_t)r * CDIM + c0 + i], alpha * (float)vv[i]);
}

// ---------------- gate edges ----------------
__global__ void k_gate(const _Float16* __restrict__ qg, const _Float16* __restrict__ kg,
                       const _Float16* __restrict__ wg, const float* __restrict__ cut,
                       const float* __restrict__ sh, const float* __restrict__ Wdeg,
                       const int* __restrict__ snd, const int* __restrict__ rcv,
                       float* __restrict__ chi) {
  int e = blockIdx.x * blockDim.y + threadIdx.y;
  if (e >= NEDGES) return;
  int lane = threadIdx.x;
  int s = snd[e], r = rcv[e];
  int c0 = lane * 4;
  const v4h qv = *(const v4h*)(qg + (size_t)r * CDIM + c0);
  const v4h kv = *(const v4h*)(kg + (size_t)s * CDIM + c0);
  const v4h wv = *(const v4h*)(wg + (size_t)e * CDIM + c0);
  float t[4];
  #pragma unroll
  for (int i = 0; i < 4; ++i) t[i] = (float)qv[i] * (float)kv[i] * (float)wv[i];
  float cj[3];
  #pragma unroll
  for (int j = 0; j < 3; ++j)
    cj[j] = t[0] * Wdeg[(c0 + 0) * 3 + j] + t[1] * Wdeg[(c0 + 1) * 3 + j] +
            t[2] * Wdeg[(c0 + 2) * 3 + j] + t[3] * Wdeg[(c0 + 3) * 3 + j];
  #pragma unroll
  for (int off = 16; off >= 1; off >>= 1) {
    cj[0] += __shfl_xor(cj[0], off, 32);
    cj[1] += __shfl_xor(cj[1], off, 32);
    cj[2] += __shfl_xor(cj[2], off, 32);
  }
  float scale = 0.088388348f * cut[e] * INV_AVG;  // 1/sqrt(C)
  if (lane < SHD) {
    int d = (lane < 3) ? 0 : (lane < 8 ? 1 : 2);
    float val = cj[d] * scale * sh[(size_t)e * SHD + lane];
    atomicAdd(&chi[(size_t)r * SHD + lane], val);
  }
}

// ---------------- concat(nf, deg_norms(chi)) -> f16 N x 160 zero padded ----------------
__global__ void k_cat16(const float* __restrict__ nf, const float* __restrict__ chi,
                        _Float16* __restrict__ cat) {
  int i = blockIdx.x * blockDim.x + threadIdx.x;
  if (i >= NNODES * CATP) return;
  int n = i / CATP, c = i % CATP;
  float v = 0.f;
  if (c < CDIM) {
    v = nf[n * CDIM + c];
  } else if (c < 131) {
    int d = c - CDIM;
    int s0  = (d == 0) ? 0 : (d == 1) ? 3 : 8;
    int cnt = (d == 0) ? 3 : (d == 1) ? 5 : 7;
    float acc = 1e-12f;
    for (int t = 0; t < cnt; ++t) { float x = chi[n * SHD + s0 + t]; acc += x * x; }
    v = sqrtf(acc);
  }
  cat[i] = (_Float16)v;
}

// ---------------- nf += h[:, :128]; chi *= (1 + expand_deg(h[:, 128:131])) ----------------
__global__ void k_inner(const float* __restrict__ h, float* __restrict__ nf,
                        float* __restrict__ chi) {
  int n = blockIdx.x, t = threadIdx.x;
  if (t < CDIM) {
    nf[n * CDIM + t] += h[(size_t)n * CATS + t];
  } else if (t < CDIM + SHD) {
    int s = t - CDIM;
    int d = (s < 3) ? 0 : (s < 8 ? 1 : 2);
    chi[n * SHD + s] *= (1.f + h[(size_t)n * CATS + CDIM + d]);
  }
}

// ---------------- final readout ----------------
__global__ void k_out(const float* __restrict__ eb, const float* __restrict__ eW2,
                      const float* __restrict__ eb2, float* __restrict__ out) {
  int n = blockIdx.x * blockDim.y + threadIdx.y;
  if (n >= NNODES) return;
  int lane = threadIdx.x, c0 = lane * 4;
  const float4 ev = *(const float4*)(eb + (size_t)n * CDIM + c0);
  float dot = ev.x * eW2[c0] + ev.y * eW2[c0 + 1] + ev.z * eW2[c0 + 2] +
              ev.w * eW2[c0 + 3];
  #pragma unroll
  for (int off = 16; off >= 1; off >>= 1) dot += __shfl_xor(dot, off, 32);
  if (lane == 0) out[n] = dot + eb2[0];
}

// =====================================================================
extern "C" void kernel_launch(void* const* d_in, const int* in_sizes, int n_in,
                              void* d_out, int out_size, void* d_ws, size_t ws_size,
                              hipStream_t stream) {
  (void)in_sizes; (void)n_in; (void)out_size; (void)ws_size;
  const float* edge_vectors = (const float*)d_in[0];
  const float* distances    = (const float*)d_in[1];
  const float* cutoffs      = (const float*)d_in[2];
  const int*   species      = (const int*)d_in[3];
  const int*   senders      = (const int*)d_in[4];
  const int*   receivers    = (const int*)d_in[5];
  // params flattened as jax pytree (dict keys sorted at each level):
  // eW1, eW2, eb1, eb2, embed, layers[l]{fb{Wk,Wo,Wq,Wv,rad_Ws[2],rad_bs[2],
  // (sph_Ws[2],sph_bs[2] l=1)}, gb{Wdeg,Wk,Wq,rad_Ws[2],rad_bs[2],(sph.. l=1)}, ib{W,b}}
  int pi = 6;
  const float* eW1   = (const float*)d_in[pi++];
  const float* eW2   = (const float*)d_in[pi++];
  const float* eb1   = (const float*)d_in[pi++];
  const float* eb2   = (const float*)d_in[pi++];
  const float* embed = (const float*)d_in[pi++];
  struct LayerP {
    const float *fWk, *fWo, *fWq, *fWv, *frW1, *frW2, *frb1, *frb2;
    const float *fsW1, *fsW2, *fsb1, *fsb2;
    const float *gWdeg, *gWk, *gWq, *grW1, *grW2, *grb1, *grb2;
    const float *gsW1, *gsW2, *gsb1, *gsb2;
    const float *ibW, *ibB;
  } L[2];
  for (int l = 0; l < 2; ++l) {
    L[l].fWk  = (const float*)d_in[pi++];
    L[l].fWo  = (const float*)d_in[pi++];
    L[l].fWq  = (const float*)d_in[pi++];
    L[l].fWv  = (const float*)d_in[pi++];
    L[l].frW1 = (const float*)d_in[pi++];
    L[l].frW2 = (const float*)d_in[pi++];
    L[l].frb1 = (const float*)d_in[pi++];
    L[l].frb2 = (const float*)d_in[pi++];
    if (l > 0) {
      L[l].fsW1 = (const float*)d_in[pi++];
      L[l].fsW2 = (const float*)d_in[pi++];
      L[l].fsb1 = (const float*)d_in[pi++];
      L[l].fsb2 = (const float*)d_in[pi++];
    } else {
      L[l].fsW1 = L[l].fsW2 = L[l].fsb1 = L[l].fsb2 = nullptr;
    }
    L[l].gWdeg = (const float*)d_in[pi++];
    L[l].gWk   = (const float*)d_in[pi++];
    L[l].gWq   = (const float*)d_in[pi++];
    L[l].grW1  = (const float*)d_in[pi++];
    L[l].grW2  = (const float*)d_in[pi++];
    L[l].grb1  = (const float*)d_in[pi++];
    L[l].grb2  = (const float*)d_in[pi++];
    if (l > 0) {
      L[l].gsW1 = (const float*)d_in[pi++];
      L[l].gsW2 = (const float*)d_in[pi++];
      L[l].gsb1 = (const float*)d_in[pi++];
      L[l].gsb2 = (const float*)d_in[pi++];
    } else {
      L[l].gsW1 = L[l].gsW2 = L[l].gsb1 = L[l].gsb2 = nullptr;
    }
    L[l].ibW = (const float*)d_in[pi++];
    L[l].ibB = (const float*)d_in[pi++];
  }

  // ---- workspace: f32 region then f16 region (all counts multiples of 8 elems)
  float* pf = (float*)d_ws;
  auto takef = [&](size_t n) { float* r = pf; pf += n; return r; };
  float* nf   = takef((size_t)NNODES * CDIM);
  float* agg  = takef((size_t)NNODES * CDIM);
  float* ebuf = takef((size_t)NNODES * CDIM);
  float* hb   = takef((size_t)NNODES * CATS);
  float* chi  = takef((size_t)NNODES * SHD + 8);
  float* shb  = takef((size_t)NEDGES * SHD + 8);
  _Float16* ph = (_Float16*)pf;
  auto takeh = [&](size_t n) { _Float16* r = ph; ph += n; return r; };
  _Float16* act16   = takeh((size_t)NNODES * CATP);
  _Float16* qb16    = takeh((size_t)NNODES * CDIM);
  _Float16* kb16    = takeh((size_t)NNODES * CDIM);
  _Float16* vb16    = takeh((size_t)NNODES * CDIM);
  _Float16* qgb16   = takeh((size_t)NNODES * CDIM);
  _Float16* kgb16   = takeh((size_t)NNODES * CDIM);
  _Float16* feats16 = takeh((size_t)NEDGES * RBFN);
  _Float16* chis16  = takeh((size_t)NEDGES * RBFN);
  _Float16* wfb16   = takeh((size_t)NEDGES * CDIM);
  _Float16* wgb16   = takeh((size_t)NEDGES * CDIM);
  _Float16* eW1t    = takeh(128 * 128);
  _Float16* Wt[2][7];     // fWq,fWk,fWv,fWo,gWq,gWk (128x128), ibW (160x160)
  _Float16* Mt[2][8];     // frW1t,frW2t,grW1t,grW2t,fsW1t,fsW2t,gsW1t,gsW2t
  for (int l = 0; l < 2; ++l) {
    for (int j = 0; j < 6; ++j) Wt[l][j] = takeh(128 * 128);
    Wt[l][6] = takeh(CATP * CATP);
    Mt[l][0] = takeh(64 * 32);  Mt[l][1] = takeh(128 * 64);
    Mt[l][2] = takeh(64 * 32);  Mt[l][3] = takeh(128 * 64);
    Mt[l][4] = takeh(64 * 32);  Mt[l][5] = takeh(128 * 64);
    Mt[l][6] = takeh(64 * 32);  Mt[l][7] = takeh(128 * 64);
  }

  auto wt = [&](const float* W, _Float16* O, int K, int N, int Kp, int Np) {
    k_wt<<<(Np * Kp + 255) / 256, 256, 0, stream>>>(W, O, K, N, Kp, Np);
  };
  auto gemm = [&](const _Float16* A, int lda, const _Float16* Bt, int ldb,
                  const float* bias, float* Cc, int ldc, int M, int Npad, int Nreal,
                  int K, int flags) {
    dim3 grid(Npad / 32, (M / 16 + 3) / 4), block(32, 4);
    k_gemm<<<grid, block, 0, stream>>>(A, lda, Bt, ldb, bias, Cc, ldc, M, Nreal, K,
                                       flags);
  };
  auto cvt = [&](const float* a, _Float16* b, int n) {
    k_cvt<<<(n + 255) / 256, 256, 0, stream>>>(a, b, n);
  };
  auto edge_mlp = [&](const _Float16* X, const _Float16* W1t, const float* b1,
                      const _Float16* W2t, const float* b2, _Float16* o, int mul) {
    k_edge_mlp<<<dim3(NEDGES / 64), dim3(32, 4), 0, stream>>>(X, W1t, b1, W2t, b2, o,
                                                              mul);
  };
  const dim3 eblk(32, 8), egrd((NEDGES + 7) / 8);

  // ---- weight prep (transpose + f16 + pad) ----
  wt(eW1, eW1t, 128, 128, 128, 128);
  for (int l = 0; l < 2; ++l) {
    const float* src[6] = {L[l].fWq, L[l].fWk, L[l].fWv, L[l].fWo, L[l].gWq, L[l].gWk};
    for (int j = 0; j < 6; ++j) wt(src[j], Wt[l][j], 128, 128, 128, 128);
    wt(L[l].ibW, Wt[l][6], 131, 131, CATP, CATP);
    wt(L[l].frW1, Mt[l][0], RBFN, HIDD, 32, 64);
    wt(L[l].frW2, Mt[l][1], HIDD, CDIM, 64, 128);
    wt(L[l].grW1, Mt[l][2], RBFN, HIDD, 32, 64);
    wt(L[l].grW2, Mt[l][3], HIDD, CDIM, 64, 128);
    if (l > 0) {
      wt(L[l].fsW1, Mt[l][4], 3, HIDD, 32, 64);
      wt(L[l].fsW2, Mt[l][5], HIDD, CDIM, 64, 128);
      wt(L[l].gsW1, Mt[l][6], 3, HIDD, 32, 64);
      wt(L[l].gsW2, Mt[l][7], HIDD, CDIM, 64, 128);
    }
  }

  k_zero<<<(NNODES * SHD + 255) / 256, 256, 0, stream>>>(chi, NNODES * SHD);
  k_embed<<<(NNODES * CDIM + 255) / 256, 256, 0, stream>>>(species, embed, nf);
  k_edge_geom<<<(NEDGES + 127) / 128, 128, 0, stream>>>(edge_vectors, distances,
                                                        feats16, shb);
  for (int l = 0; l < 2; ++l) {
    if (l > 0)
      k_chi_scalar<<<(NEDGES + 127) / 128, 128, 0, stream>>>(chi, senders, receivers,
                                                             chis16);
    // feature block
    cvt(nf, act16, NNODES * CDIM);
    gemm(act16, CDIM, Wt[l][0], 128, nullptr, (float*)qb16, CDIM, NNODES, 128, 128,
         128, 8);
    gemm(act16, CDIM, Wt[l][1], 128, nullptr, (float*)kb16, CDIM, NNODES, 128, 128,
         128, 8);
    gemm(act16, CDIM, Wt[l][2], 128, nullptr, (float*)vb16, CDIM, NNODES, 128, 128,
         128, 8);
    edge_mlp(feats16, Mt[l][0], L[l].frb1, Mt[l][1], L[l].frb2, wfb16, 0);
    if (l > 0)
      edge_mlp(chis16, Mt[l][4], L[l].fsb1, Mt[l][5], L[l].fsb2, wfb16, 1);
    k_zero<<<(NNODES * CDIM + 255) / 256, 256, 0, stream>>>(agg, NNODES * CDIM);
    k_attn<<<egrd, eblk, 0, stream>>>(qb16, kb16, vb16, wfb16, cutoffs, senders,
                                      receivers, agg);
    cvt(agg, act16, NNODES * CDIM);
    gemm(act16, CDIM, Wt[l][3], 128, nullptr, nf, CDIM, NNODES, 128, 128, 128, 2);
    // gate block
    cvt(nf, act16, NNODES * CDIM);
    gemm(act16, CDIM, Wt[l][4], 128, nullptr, (float*)qgb16, CDIM, NNODES, 128, 128,
         128, 8);
    gemm(act16, CDIM, Wt[l][5], 128, nullptr, (float*)kgb16, CDIM, NNODES, 128, 128,
         128, 8);
    edge_mlp(feats16, Mt[l][2], L[l].grb1, Mt[l][3], L[l].grb2, wgb16, 0);
    if (l > 0)
      edge_mlp(chis16, Mt[l][6], L[l].gsb1, Mt[l][7], L[l].gsb2, wgb16, 1);
    k_gate<<<egrd, eblk, 0, stream>>>(qgb16, kgb16, wgb16, cutoffs, shb, L[l].gWdeg,
                                      senders, receivers, chi);
    // interaction block
    k_cat16<<<(NNODES * CATP + 255) / 256, 256, 0, stream>>>(nf, chi, act16);
    gemm(act16, CATP, Wt[l][6], CATP, L[l].ibB, hb, CATS, NNODES, CATP, 131, CATP, 4);
    k_inner<<<NNODES, 160, 0, stream>>>(hb, nf, chi);
  }
  cvt(nf, act16, NNODES * CDIM);
  gemm(act16, CDIM, eW1t, 128, eb1, ebuf, CDIM, NNODES, 128, 128, 128, 5);
  k_out<<<dim3((NNODES + 7) / 8), dim3(32, 8), 0, stream>>>(ebuf, eW2, eb2,
                                                            (float*)d_out);
}
